// AttentionGated_71451075937077
// MI455X (gfx1250) — matrix operand order repaired and checked by reading, not compiled, not run on hardware
//
#include <hip/hip_runtime.h>
#include <hip/hip_bf16.h>

typedef __bf16 bf16_t;
typedef __attribute__((ext_vector_type(16))) __bf16 v16bf;
typedef __attribute__((ext_vector_type(8)))  __bf16 v8bf;
typedef __attribute__((ext_vector_type(4)))  __bf16 v4bf;
typedef __attribute__((ext_vector_type(2)))  __bf16 v2bf;
typedef __attribute__((ext_vector_type(8)))  float   v8f;

#define D_IN    1024
#define L_DIM   512
#define D_DIM   384
#define N_INST  40000
#define BATCH   2
#define ROWS    (BATCH * N_INST)   // 80000
#define MT      32                 // instance rows per workgroup (pass1)
#define XSTR    1032               // padded LDS stride for x tile (bf16 elems)
#define HSTR    520                // padded LDS stride for h tile (bf16 elems)
#define NT_F    (L_DIM / 16)       // 32 N-tiles for Wf
#define KT_F    (D_IN / 32)        // 32 K-tiles for Wf
#define NT_A    (D_DIM / 16)       // 24 N-tiles for Wa/Wb
#define KT_A    (L_DIM / 32)       // 16 K-tiles for Wa/Wb
#define CHUNKS3 160                // pass3 row chunks (40000/160 = 250)
#define BSLICE  (NT_F * 512)       // 16384 bf16 = 32KB k-slice of packed Wf

// ---------------------------------------------------------------------------
// CDNA5 async global->LDS copy (ASYNCcnt-tracked) via inline asm.
// GV addressing form: global_load_async_to_lds_b128 vDstLds, v[addr64], off
// Host pass parses but never emits this (device-only body).
// ---------------------------------------------------------------------------
__device__ __forceinline__ void async_copy_b128(const void* g, void* l) {
#if defined(__AMDGCN__)
    asm volatile("global_load_async_to_lds_b128 %0, %1, off"
                 :: "v"((unsigned)(unsigned long long)l),
                    "v"((unsigned long long)g)
                 : "memory");
#else
    (void)g; (void)l;
#endif
}

__device__ __forceinline__ void async_wait0() {
#if defined(__AMDGCN__)
    asm volatile("s_wait_asynccnt 0x0" ::: "memory");
#endif
}

// ---------------------------------------------------------------------------
// Prep: repack Wf/Wa/Wb (fp32 [K,N] row-major) into bf16 WMMA-B tile layout.
// Tile (kt,nt) = 512 bf16: element [lane*16 + j] = W[kt*32 + (lane/16)*16 + j][nt*16 + lane%16]
// so each lane's B fragment is one contiguous 32-byte v16bf load.
// ---------------------------------------------------------------------------
__global__ __launch_bounds__(256) void prep_kernel(
    const float* __restrict__ Wf, const float* __restrict__ Wa, const float* __restrict__ Wb,
    bf16_t* __restrict__ wfP, bf16_t* __restrict__ waP, bf16_t* __restrict__ wbP)
{
    const int idx    = blockIdx.x * blockDim.x + threadIdx.x;
    const int stride = gridDim.x * blockDim.x;

    for (int e = idx; e < D_IN * L_DIM; e += stride) {
        int tileI = e >> 9;
        int r     = e & 511;
        int lane  = r >> 4, j = r & 15;
        int kt    = tileI >> 5;       // /NT_F(=32)
        int nt    = tileI & 31;
        int k     = kt * 32 + (lane >> 4) * 16 + j;
        int n     = nt * 16 + (lane & 15);
        wfP[e] = (bf16_t)Wf[k * L_DIM + n];
    }
    for (int e = idx; e < L_DIM * D_DIM; e += stride) {
        int tileI = e >> 9;
        int r     = e & 511;
        int lane  = r >> 4, j = r & 15;
        int kt    = tileI / NT_A;
        int nt    = tileI % NT_A;
        int k     = kt * 32 + (lane >> 4) * 16 + j;
        int n     = nt * 16 + (lane & 15);
        waP[e] = (bf16_t)Wa[k * D_DIM + n];
        wbP[e] = (bf16_t)Wb[k * D_DIM + n];
    }
}

// ---------------------------------------------------------------------------
// Pass 1: per 32-row tile, fused  h = relu(x@Wf+bf);  s = (tanh(h@Wa+ba)*
// sigmoid(h@Wb+bb)) . Wc + bc.  Writes h (bf16) and s (f32) to global.
// Stage F double-buffers 32KB Wf k-slices into LDS with async DMA.
// h-tile LDS aliases the x-tile LDS (x dead after the stage-F k-loop).
// ---------------------------------------------------------------------------
__global__ __launch_bounds__(256) void pass1_kernel(
    const float* __restrict__ x,  const float* __restrict__ bfv,
    const float* __restrict__ bav, const float* __restrict__ bbv,
    const float* __restrict__ wcv, const float* __restrict__ bcv,
    const bf16_t* __restrict__ wfP, const bf16_t* __restrict__ waP,
    const bf16_t* __restrict__ wbP,
    bf16_t* __restrict__ hOut, float* __restrict__ sOut)
{
    __shared__ __align__(32) bf16_t xh[MT * XSTR];        // x tile, later reused as h tile
    __shared__ __align__(32) bf16_t bstage[2][BSLICE];    // 2 x 32KB Wf k-slices
    __shared__ float bfS[L_DIM];
    __shared__ float baS[D_DIM], bbS[D_DIM], wcS[D_DIM];
    __shared__ float sAcc[MT];

    const int tid   = threadIdx.x;
    const int wave  = tid >> 5;
    const int lane  = tid & 31;
    const int lhalf = lane >> 4;      // 0: K 0-15, 1: K 16-31
    const int lmod  = lane & 15;
    const int mgrp  = wave >> 2;      // 0..1 -> 16-row half
    const int ngrp  = wave & 3;       // 0..3 -> column strip
    const size_t rowBase = (size_t)blockIdx.x * MT;

    for (int i = tid; i < L_DIM; i += 256) bfS[i] = bfv[i];
    for (int i = tid; i < D_DIM; i += 256) { baS[i] = bav[i]; bbS[i] = bbv[i]; wcS[i] = wcv[i]; }
    if (tid < MT) sAcc[tid] = 0.f;

    // Stage x tile (32 x 1024 fp32) into LDS as bf16, padded stride.
    const float4* xv = (const float4*)(x + rowBase * D_IN);
    for (int i = tid; i < MT * (D_IN / 4); i += 256) {
        int r  = i >> 8;
        int c4 = i & 255;
        float4 f = xv[i];
        v4bf bvec = { (bf16_t)f.x, (bf16_t)f.y, (bf16_t)f.z, (bf16_t)f.w };
        *(v4bf*)&xh[r * XSTR + c4 * 4] = bvec;
    }

    // Async-stage first Wf k-slice: 32KB contiguous, 128B (8 x b128) per thread.
    const char* wfBytes = (const char*)wfP;
    {
        const char* g = wfBytes + (size_t)tid * 128;
        bf16_t*     l = &bstage[0][0] + tid * 64;
        #pragma unroll
        for (int j = 0; j < 8; ++j)
            async_copy_b128(g + j * 16, l + j * 8);
    }
    async_wait0();
    __syncthreads();

    // ---- Stage F: h = relu(x @ Wf + bf) ----
    v8f acc[8];
    {
        v8f z = {0.f,0.f,0.f,0.f,0.f,0.f,0.f,0.f};
        #pragma unroll
        for (int t = 0; t < 8; ++t) acc[t] = z;
    }
    const bf16_t* arow = &xh[(mgrp * 16 + lmod) * XSTR];
    const int ntBase = ngrp * 8;      // 8 column tiles (128 cols) per wave

    for (int kt = 0; kt < KT_F; ++kt) {
        const int cur = kt & 1;
        if (kt + 1 < KT_F) {          // DMA next slice into the other buffer
            const char* g = wfBytes + (size_t)(kt + 1) * (BSLICE * 2) + tid * 128;
            bf16_t*     l = &bstage[cur ^ 1][0] + tid * 64;
            #pragma unroll
            for (int j = 0; j < 8; ++j)
                async_copy_b128(g + j * 16, l + j * 8);
        }

        const int k0 = kt * 32;
        union { v16bf v; v8bf h2[2]; } af;
        af.h2[0] = *(const v8bf*)&arow[k0 + lhalf * 16];
        af.h2[1] = *(const v8bf*)&arow[k0 + lhalf * 16 + 8];

        const bf16_t* bs = &bstage[cur][0];
        v16bf bw[8];
        #pragma unroll
        for (int t = 0; t < 8; ++t)
            bw[t] = *(const v16bf*)&bs[(ntBase + t) * 512 + lane * 16];
        #pragma unroll
        for (int t = 0; t < 8; ++t)
            acc[t] = __builtin_amdgcn_wmma_f32_16x16x32_bf16(
                false, af.v, false, bw[t], (short)0, acc[t], false, false);

        async_wait0();                // next slice fully resident in LDS
        __syncthreads();              // everyone done reading/writing buffers
    }

    // Epilogue: bias + relu -> h tile (bf16), aliased onto the x-tile LDS.
    bf16_t* const hs = xh;            // x is dead past the barrier above
    #pragma unroll
    for (int t = 0; t < 8; ++t) {
        const int col  = (ntBase + t) * 16 + lmod;
        const float bb = bfS[col];
        #pragma unroll
        for (int i = 0; i < 8; ++i) {
            float v = acc[t][i] + bb;
            v = v > 0.f ? v : 0.f;
            const int m = mgrp * 16 + i + 8 * lhalf;
            hs[m * HSTR + col] = (bf16_t)v;
        }
    }
    __syncthreads();

    // Coalesced bf16 store of the h tile to HBM (for pass 3).
    for (int i = tid; i < MT * (L_DIM / 8); i += 256) {
        int r  = i >> 6;
        int c8 = i & 63;
        v8bf hv = *(const v8bf*)&hs[r * HSTR + c8 * 8];
        *(v8bf*)&hOut[(rowBase + r) * L_DIM + c8 * 8] = hv;
    }

    // ---- Stage A/G: a = tanh(h@Wa+ba), g = sigmoid(h@Wb+bb), s = (a*g).Wc ----
    v8f aacc[6], gacc[6];
    {
        v8f z = {0.f,0.f,0.f,0.f,0.f,0.f,0.f,0.f};
        #pragma unroll
        for (int t = 0; t < 6; ++t) { aacc[t] = z; gacc[t] = z; }
    }
    const bf16_t* hrow = &hs[(mgrp * 16 + lmod) * HSTR];
    const int ntB2 = ngrp * 6;        // 6 column tiles (96 cols) per wave
    for (int kt = 0; kt < KT_A; ++kt) {
        const int k0 = kt * 32;
        union { v16bf v; v8bf h2[2]; } af;
        af.h2[0] = *(const v8bf*)&hrow[k0 + lhalf * 16];
        af.h2[1] = *(const v8bf*)&hrow[k0 + lhalf * 16 + 8];

        const bf16_t* ta = waP + ((size_t)kt * NT_A + ntB2) * 512;
        const bf16_t* tb = wbP + ((size_t)kt * NT_A + ntB2) * 512;
        if (kt + 1 < KT_A) {
            __builtin_prefetch(waP + ((size_t)(kt + 1) * NT_A + ntB2) * 512, 0, 1);
            __builtin_prefetch(wbP + ((size_t)(kt + 1) * NT_A + ntB2) * 512, 0, 1);
        }
        // Batch all 12 B fragments first: one load clause + one wait,
        // then an uninterrupted chain of 12 WMMAs.
        v16bf bwa[6], bwb[6];
        #pragma unroll
        for (int t = 0; t < 6; ++t) {
            bwa[t] = *(const v16bf*)&ta[t * 512 + lane * 16];
            bwb[t] = *(const v16bf*)&tb[t * 512 + lane * 16];
        }
        #pragma unroll
        for (int t = 0; t < 6; ++t) {
            aacc[t] = __builtin_amdgcn_wmma_f32_16x16x32_bf16(
                false, af.v, false, bwa[t], (short)0, aacc[t], false, false);
            gacc[t] = __builtin_amdgcn_wmma_f32_16x16x32_bf16(
                false, af.v, false, bwb[t], (short)0, gacc[t], false, false);
        }
    }
    float part[8] = {0.f,0.f,0.f,0.f,0.f,0.f,0.f,0.f};
    #pragma unroll
    for (int t = 0; t < 6; ++t) {
        const int col = (ntB2 + t) * 16 + lmod;
        const float wcc = wcS[col], bac = baS[col], bbc = bbS[col];
        #pragma unroll
        for (int i = 0; i < 8; ++i) {
            float av = tanhf(aacc[t][i] + bac);
            float gv = 1.f / (1.f + __expf(-(gacc[t][i] + bbc)));
            part[i] += av * gv * wcc;
        }
    }
    #pragma unroll
    for (int i = 0; i < 8; ++i)
        atomicAdd(&sAcc[mgrp * 16 + i + 8 * lhalf], part[i]);
    __syncthreads();
    if (tid < MT) sOut[rowBase + tid] = sAcc[tid] + bcv[0];
}

// ---------------------------------------------------------------------------
// Pass 2: per-batch softmax statistics (max, sum of exp).
// ---------------------------------------------------------------------------
__global__ __launch_bounds__(256) void pass2_kernel(
    const float* __restrict__ s, float* __restrict__ stats)
{
    __shared__ float red[256];
    const int b   = blockIdx.x;
    const int tid = threadIdx.x;
    const float* sb = s + (size_t)b * N_INST;

    float mx = -3.402823466e38f;
    for (int i = tid; i < N_INST; i += 256) mx = fmaxf(mx, sb[i]);
    red[tid] = mx; __syncthreads();
    for (int o = 128; o > 0; o >>= 1) {
        if (tid < o) red[tid] = fmaxf(red[tid], red[tid + o]);
        __syncthreads();
    }
    mx = red[0]; __syncthreads();

    float sum = 0.f;
    for (int i = tid; i < N_INST; i += 256) sum += __expf(sb[i] - mx);
    red[tid] = sum; __syncthreads();
    for (int o = 128; o > 0; o >>= 1) {
        if (tid < o) red[tid] += red[tid + o];
        __syncthreads();
    }
    if (tid == 0) { stats[b * 2 + 0] = mx; stats[b * 2 + 1] = red[0]; }
}

// ---------------------------------------------------------------------------
// Pass 3: partial weighted reductions  partial[chunk][b][l] = sum_n w_n h[n][l]
// (deterministic: no atomics). Pass 3b: reduce chunks into pooled[b][l].
// ---------------------------------------------------------------------------
__global__ __launch_bounds__(256) void pass3_kernel(
    const bf16_t* __restrict__ h, const float* __restrict__ s,
    const float* __restrict__ stats, float* __restrict__ partial)
{
    const int b     = blockIdx.y;
    const int chunk = blockIdx.x;
    const int c     = threadIdx.x * 2;
    const float mx  = stats[b * 2];
    const bf16_t* hb = h + ((size_t)b * N_INST) * L_DIM;
    const float*  sb = s + (size_t)b * N_INST;
    const int n0 = chunk * (N_INST / CHUNKS3);

    float a0 = 0.f, a1 = 0.f;
    for (int r = 0; r < N_INST / CHUNKS3; ++r) {
        const int n = n0 + r;
        const float w = __expf(sb[n] - mx);
        v2bf p = *(const v2bf*)&hb[(size_t)n * L_DIM + c];
        a0 += w * (float)p[0];
        a1 += w * (float)p[1];
    }
    float* dst = partial + ((size_t)chunk * BATCH + b) * L_DIM;
    dst[c] = a0; dst[c + 1] = a1;
}

__global__ __launch_bounds__(256) void pass3b_kernel(
    const float* __restrict__ partial, float* __restrict__ pooled)
{
    const int b = blockIdx.x;
    for (int c = threadIdx.x; c < L_DIM; c += 256) {
        float acc = 0.f;
        for (int ch = 0; ch < CHUNKS3; ++ch)
            acc += partial[((size_t)ch * BATCH + b) * L_DIM + c];
        pooled[b * L_DIM + c] = acc;
    }
}

// ---------------------------------------------------------------------------
// Pass 4: logits[b][c] = (pooled[b]/Z_b) @ Wcls[:,c] + bcls[c]
// ---------------------------------------------------------------------------
__global__ __launch_bounds__(128) void pass4_kernel(
    const float* __restrict__ pooled, const float* __restrict__ stats,
    const float* __restrict__ Wcls, const float* __restrict__ bcls,
    float* __restrict__ out)
{
    const int tid  = threadIdx.x;        // 128 = 4 groups x 32 lanes
    const int g    = tid >> 5;
    const int lane = tid & 31;
    const int b = g >> 1, c = g & 1;
    const float z = stats[b * 2 + 1];
    float acc = 0.f;
    for (int l = lane; l < L_DIM; l += 32)
        acc += (pooled[b * L_DIM + l] / z) * Wcls[l * 2 + c];
    for (int m = 16; m > 0; m >>= 1)
        acc += __shfl_xor(acc, m, 32);
    if (lane == 0) out[b * 2 + c] = acc + bcls[c];
}

// ---------------------------------------------------------------------------
extern "C" void kernel_launch(void* const* d_in, const int* in_sizes, int n_in,
                              void* d_out, int out_size, void* d_ws, size_t ws_size,
                              hipStream_t stream) {
    const float* x    = (const float*)d_in[0];
    const float* Wf   = (const float*)d_in[1];
    const float* bfv  = (const float*)d_in[2];
    const float* Wa   = (const float*)d_in[3];
    const float* bav  = (const float*)d_in[4];
    const float* Wb   = (const float*)d_in[5];
    const float* bbv  = (const float*)d_in[6];
    const float* Wc   = (const float*)d_in[7];
    const float* bcv  = (const float*)d_in[8];
    const float* Wcls = (const float*)d_in[9];
    const float* bcls = (const float*)d_in[10];
    float* out = (float*)d_out;

    char*  ws  = (char*)d_ws;
    size_t off = 0;
    auto alloc = [&](size_t bytes) {
        void* p = ws + off;
        off = (off + bytes + 255) & ~(size_t)255;
        return p;
    };
    bf16_t* hbuf    = (bf16_t*)alloc((size_t)ROWS * L_DIM * 2);
    float*  sbuf    = (float*) alloc((size_t)ROWS * 4);
    float*  stats   = (float*) alloc(4 * 4);
    float*  pooled  = (float*) alloc((size_t)BATCH * L_DIM * 4);
    float*  partial = (float*) alloc((size_t)CHUNKS3 * BATCH * L_DIM * 4);
    bf16_t* wfP     = (bf16_t*)alloc((size_t)D_IN * L_DIM * 2);
    bf16_t* waP     = (bf16_t*)alloc((size_t)L_DIM * D_DIM * 2);
    bf16_t* wbP     = (bf16_t*)alloc((size_t)L_DIM * D_DIM * 2);

    prep_kernel <<<512, 256, 0, stream>>>(Wf, Wa, Wb, wfP, waP, wbP);
    pass1_kernel<<<ROWS / MT, 256, 0, stream>>>(x, bfv, bav, bbv, Wc, bcv,
                                                wfP, waP, wbP, hbuf, sbuf);
    pass2_kernel<<<BATCH, 256, 0, stream>>>(sbuf, stats);
    pass3_kernel<<<dim3(CHUNKS3, BATCH), 256, 0, stream>>>(hbuf, sbuf, stats, partial);
    pass3b_kernel<<<BATCH, 256, 0, stream>>>(partial, pooled);
    pass4_kernel<<<1, 128, 0, stream>>>(pooled, stats, Wcls, bcls, out);
}
